// DecorrelationLoss_27633819582907
// MI455X (gfx1250) — matrix operand order repaired and checked by reading, not compiled, not run on hardware
//
#include <hip/hip_runtime.h>

typedef __attribute__((ext_vector_type(16))) __bf16 v16bf;
typedef __attribute__((ext_vector_type(8)))  __bf16 v8bf;
typedef __attribute__((ext_vector_type(8)))  float  v8f;
typedef __attribute__((ext_vector_type(4)))  unsigned int v4u;
typedef __attribute__((ext_vector_type(8)))  int    v8i;
typedef __attribute__((ext_vector_type(4)))  int    v4i;

#define B_DIM  8
#define C_DIM  1024
#define K_DIM  4096
#define BM     128
#define BN     128
#define KC     64
#define NC     (K_DIM / KC)          /* 64 K-chunks */
#define NSPLIT (C_DIM / BN)          /* 8 col-block partials per row */
#define GPITCH 132                   /* floats; 128+4 pad -> conflict-free */
#define GST_BYTES  (BM * GPITCH * 4) /* 67584 */
#define STAGE_BYTES 16384            /* one staged region: 8 groups x 2KB */
#define SMEM_BYTES (GST_BYTES + 4 * STAGE_BYTES)  /* Gst + 2x(A,B) = 133120 */
#define GRP_BYTES  131072            /* 128 ksubs x 1KB per 16-row group */
#define BATCH_OP_BYTES ((size_t)64 * GRP_BYTES)   /* 8 MB per batch */
#define OP_TOTAL_BYTES ((size_t)B_DIM * BATCH_OP_BYTES) /* 64 MB */
#define SCALE  5.0f                  /* 1 / TEMPERATURE */
#define EPS    1e-10f

// fp32 -> bf16 round-to-nearest-even, two values packed into one dword
__device__ __forceinline__ unsigned pack2_bf16(float a, float b) {
  unsigned ua = __float_as_uint(a), ub = __float_as_uint(b);
  ua = (ua + 0x7FFFu + ((ua >> 16) & 1u)) >> 16;
  ub = (ub + 0x7FFFu + ((ub >> 16) & 1u)) >> 16;
  return (ua & 0xFFFFu) | (ub << 16);
}

// Issue one TDM 2-D tile load: 8 rows ("groups") x 2048B, row stride 128KB,
// packed contiguously into LDS at ldsoff. data_size = 8B.
__device__ __forceinline__ void tdm_load(unsigned long long gaddr,
                                         unsigned ldsoff) {
  v4u g0 = {1u,                                   // count=1 descriptor
            ldsoff,                               // lds_addr (bytes)
            (unsigned)gaddr,                      // global_addr[31:0]
            (unsigned)((gaddr >> 32) & 0x01FFFFFFu) | (2u << 30)}; // +type=2
  v8i g1 = {(int)(3u << 16),          // data_size = 3 (8B); no mask/pad/iter
            (int)0x40000000u,         // tensor_dim0[15:0]=0x4000 -> 16384
            0,                        // tensor_dim0 hi / tensor_dim1 lo
            (int)0x01000001u,         // tensor_dim1 hi=1 (65536), tile_dim0=256
            8,                        // tile_dim1 = 8 rows, tile_dim2 = 0
            16384,                    // tensor_dim0_stride lo (16384 elem=128KB)
            0, 0};                    // stride hi, dim1_stride = 0
  v4i g2 = {0, 0, 0, 0};
  v4i g3 = {0, 0, 0, 0};
  v8i g4 = {0, 0, 0, 0, 0, 0, 0, 0};
  __builtin_amdgcn_tensor_load_to_lds(g0, g1, g2, g3, g4, 0);
}

// ---------------------------------------------------------------------------
// Kernel 0: fp32 -> bf16, rewritten into the WMMA A-operand tile layout.
// One WG handles one (batch, 16-row group, quarter-K) -> 32 x 1KB blocks,
// staged in LDS so the global writes are fully coalesced b128s.
// ---------------------------------------------------------------------------
__global__ __launch_bounds__(256) void convert_layout(
    const float* __restrict__ x, unsigned char* __restrict__ wsOP) {
  __shared__ __align__(16) unsigned char cbuf[32768];
  const int t = threadIdx.x;
  const int wg = blockIdx.x;               // 8 * 64 * 4 = 2048
  const int q  = wg & 3;                   // quarter of K (32 ksubs)
  const int gg = (wg >> 2) & 63;           // 16-row group
  const int b  = wg >> 8;
  const float* xb = x + (size_t)b * C_DIM * K_DIM;

  #pragma unroll 4
  for (int it = 0; it < 16; ++it) {
    int li = t + 256 * it;                 // 0..4095 float4 slots
    int m  = li >> 8;                      // row within group (16 rows)
    int kq = li & 255;                     // float4 within 1024-K quarter
    int k0 = kq * 4;
    const float4 v =
        *(const float4*)(xb + (size_t)(gg * 16 + m) * K_DIM + q * 1024 + k0);
    unsigned lo = pack2_bf16(v.x, v.y);
    unsigned hi = pack2_bf16(v.z, v.w);
    int lksub = k0 >> 5, kl = k0 & 31;
    int h = (kl >> 3) & 1, seg = kl >> 4;
    *(uint2*)(cbuf + lksub * 1024 + (m + 16 * h) * 32 + seg * 16 +
              2 * (kl & 7)) = make_uint2(lo, hi);
  }
  __syncthreads();
  unsigned char* dst =
      wsOP + ((size_t)(b * 64 + gg) * 128 + (size_t)q * 32) * 1024;
  #pragma unroll
  for (int it = 0; it < 8; ++it) {
    int off = (t + 256 * it) * 16;
    *(float4*)(dst + off) = *(const float4*)(cbuf + off);
  }
}

// ---------------------------------------------------------------------------
// Kernel 1: one 128x128 block of G = X X^T per WG. Operands staged by the
// Tensor Data Mover (double buffered), consumed by bf16 WMMA, fused partial
// row-softmax stats written to workspace.
// ---------------------------------------------------------------------------
__global__ __launch_bounds__(256) void gram_softmax_partial(
    const unsigned char* __restrict__ wsOP, float* __restrict__ wsM,
    float* __restrict__ wsS, float* __restrict__ wsD) {
  extern __shared__ unsigned char smem[];
  float* Gst = (float*)smem;               // 128 x 132 staged G block

  const int t    = threadIdx.x;
  const int lane = t & 31;
  const int w    = t >> 5;
  const int wg   = blockIdx.x;             // 8 * 8 * 8 = 512
  const int b    = wg >> 6;
  const int rblk = (wg >> 3) & 7;
  const int cblk = wg & 7;
  const int r0 = rblk * BM, c0 = cblk * BN;

  const unsigned smbase = (unsigned)(unsigned long long)(uintptr_t)smem;
  const size_t opb = (size_t)b * BATCH_OP_BYTES;
  const unsigned long long gA0 =
      (unsigned long long)(uintptr_t)(wsOP + opb + (size_t)(rblk * 8) * GRP_BYTES);
  const unsigned long long gB0 =
      (unsigned long long)(uintptr_t)(wsOP + opb + (size_t)(cblk * 8) * GRP_BYTES);

  // wave -> 2 row-groups x 4 col-groups of 16x16 tiles
  const int ia = (w & 3) * 2;
  const int jb = (w >> 2) * 4;

  v8f acc[2][4];
  #pragma unroll
  for (int r = 0; r < 2; ++r)
    #pragma unroll
    for (int j = 0; j < 4; ++j)
      acc[r][j] = (v8f){0.f, 0.f, 0.f, 0.f, 0.f, 0.f, 0.f, 0.f};

  // prologue: stage chunk 0 into buffer 0
  if (w == 0) {
    tdm_load(gA0, smbase + GST_BYTES);
    tdm_load(gB0, smbase + GST_BYTES + STAGE_BYTES);
  }

  for (int kc = 0; kc < NC; ++kc) {
    const int p = kc & 1;
    if (w == 0) {
      if (kc + 1 < NC) {
        const unsigned boff = GST_BYTES + (unsigned)(1 - p) * 2 * STAGE_BYTES;
        tdm_load(gA0 + (size_t)(2 * (kc + 1)) * 1024, smbase + boff);
        tdm_load(gB0 + (size_t)(2 * (kc + 1)) * 1024,
                 smbase + boff + STAGE_BYTES);
        __builtin_amdgcn_s_wait_tensorcnt(2);  // chunk kc's pair is done
      } else {
        __builtin_amdgcn_s_wait_tensorcnt(0);
      }
    }
    __syncthreads();

    const unsigned char* pa = smem + GST_BYTES + p * 2 * STAGE_BYTES;
    const unsigned char* pb = pa + STAGE_BYTES;
    const int boffB = (lane & 15) * 32 + (lane >> 4) * 512;
    #pragma unroll
    for (int ks = 0; ks < 2; ++ks) {
      v16bf a0 = *(const v16bf*)(pa + ia * 2048 + ks * 1024 + lane * 32);
      v16bf a1 = *(const v16bf*)(pa + (ia + 1) * 2048 + ks * 1024 + lane * 32);
      #pragma unroll
      for (int j = 0; j < 4; ++j) {
        const unsigned char* qp = pb + (jb + j) * 2048 + ks * 1024 + boffB;
        v8bf blo = *(const v8bf*)qp;          // K = 8h .. 8h+7
        v8bf bhi = *(const v8bf*)(qp + 16);   // K = 16+8h .. 16+8h+7
        v16bf bb = __builtin_shufflevector(blo, bhi, 0, 1, 2, 3, 4, 5, 6, 7, 8,
                                           9, 10, 11, 12, 13, 14, 15);
        acc[0][j] = __builtin_amdgcn_wmma_f32_16x16x32_bf16(
            false, a0, false, bb, (short)0, acc[0][j], false, false);
        acc[1][j] = __builtin_amdgcn_wmma_f32_16x16x32_bf16(
            false, a1, false, bb, (short)0, acc[1][j], false, false);
      }
    }
    __syncthreads();
  }

  // ---- stage accumulators to LDS (C/D layout: lane = n + 16*mh, VGPR v -> M)
  {
    const int nn = lane & 15, mh = lane >> 4;
    #pragma unroll
    for (int r = 0; r < 2; ++r)
      #pragma unroll
      for (int j = 0; j < 4; ++j)
        #pragma unroll
        for (int v = 0; v < 8; ++v)
          Gst[((ia + r) * 16 + mh * 8 + v) * GPITCH + (jb + j) * 16 + nn] =
              acc[r][j][v];
  }
  __syncthreads();

  // ---- partial softmax stats over this 128-column block
  if (t < BM) {
    const int gr = r0 + t;
    const float* row = Gst + t * GPITCH;
    float mx = -3.4e38f;
    for (int c = 0; c < BN; ++c) mx = fmaxf(mx, row[c] * SCALE);
    float s = 0.f;
    for (int c = 0; c < BN; ++c) s += __expf(row[c] * SCALE - mx);
    const int pi = (b * C_DIM + gr) * NSPLIT + cblk;
    wsM[pi] = mx;
    wsS[pi] = s;
    if (cblk == rblk) wsD[b * C_DIM + gr] = row[gr - c0] * SCALE;
  }
}

// ---------------------------------------------------------------------------
// Kernel 2: merge the 8 (max, sumexp) partials per row -> per-row loss
// ---------------------------------------------------------------------------
__global__ __launch_bounds__(256) void merge_rows(
    const float* __restrict__ wsM, const float* __restrict__ wsS,
    const float* __restrict__ wsD, float* __restrict__ wsL) {
  const int q = blockIdx.x * 256 + threadIdx.x;   // 0 .. 8191
  float M = -3.4e38f;
  #pragma unroll
  for (int p = 0; p < NSPLIT; ++p) M = fmaxf(M, wsM[q * NSPLIT + p]);
  float S = 0.f;
  #pragma unroll
  for (int p = 0; p < NSPLIT; ++p)
    S += wsS[q * NSPLIT + p] * __expf(wsM[q * NSPLIT + p] - M);
  const float pd = __expf(wsD[q] - M) / S;
  wsL[q] = -__logf(pd + EPS);
}

// ---------------------------------------------------------------------------
// Kernel 3: deterministic fixed-order reduction -> mean loss
// ---------------------------------------------------------------------------
__global__ __launch_bounds__(256) void final_reduce(
    const float* __restrict__ wsL, float* __restrict__ out) {
  __shared__ float sm[256];
  const int t = threadIdx.x;
  float a = 0.f;
  for (int i = t; i < B_DIM * C_DIM; i += 256) a += wsL[i];
  sm[t] = a;
  __syncthreads();
  for (int s = 128; s > 0; s >>= 1) {
    if (t < s) sm[t] += sm[t + s];
    __syncthreads();
  }
  if (t == 0) out[0] = sm[0] * (1.0f / (float)(B_DIM * C_DIM));
}

extern "C" void kernel_launch(void* const* d_in, const int* in_sizes, int n_in,
                              void* d_out, int out_size, void* d_ws,
                              size_t ws_size, hipStream_t stream) {
  (void)in_sizes; (void)n_in; (void)out_size; (void)ws_size;
  const float* x = (const float*)d_in[0];
  float* out = (float*)d_out;
  unsigned char* wsOP = (unsigned char*)d_ws;      // 64 MB bf16 operand tiles
  float* wsM = (float*)(wsOP + OP_TOTAL_BYTES);    // [8][1024][8]
  float* wsS = wsM + B_DIM * C_DIM * NSPLIT;       // [8][1024][8]
  float* wsD = wsS + B_DIM * C_DIM * NSPLIT;       // [8][1024]
  float* wsL = wsD + B_DIM * C_DIM;                // [8][1024]
  // total ws: 64 MB + 576 KB

  convert_layout<<<B_DIM * 64 * 4, 256, 0, stream>>>(x, wsOP);
  const int grid1 = B_DIM * (C_DIM / BM) * (C_DIM / BN);   // 512
  gram_softmax_partial<<<grid1, 256, SMEM_BYTES, stream>>>(wsOP, wsM, wsS, wsD);
  merge_rows<<<(B_DIM * C_DIM) / 256, 256, 0, stream>>>(wsM, wsS, wsD, wsL);
  final_reduce<<<1, 256, 0, stream>>>(wsL, out);
}
